// QGNNAgent_44908178047112
// MI455X (gfx1250) — compile-verified, matrix operand order checked by reading, not run on hardware
//
#include <hip/hip_runtime.h>
#include <hip/hip_bf16.h>
#include <math.h>

// ---------------- dims ----------------
#define BB   512
#define AA   32
#define OBS  128
#define HH   64
#define NACT 16
#define EHID 96          // H*3/2
#define QHID 104         // (OBS+H+NACT)/2
#define QPAD 112         // QHID padded to 16
#define NODES (BB*AA)    // 16384

typedef __attribute__((ext_vector_type(16))) _Float16 v16h;
typedef __attribute__((ext_vector_type(8)))  _Float16 v8h;
typedef __attribute__((ext_vector_type(8)))  float    v8f;

#define DEVINL __device__ __forceinline__

// ---------------- WMMA fragment helpers (wave32, 16x16x32 f16 -> f32) -------
// A-matrix 16x32 layout (ISA 7.12.2): lane L holds row M=L&15;
//   lanes 0-15  : elems 0..7 -> K=k0+0..7,   elems 8..15 -> K=k0+16..23
//   lanes 16-31 : elems 0..7 -> K=k0+8..15,  elems 8..15 -> K=k0+24..31
// Same pattern for B from a row-major (N x K) f16 weight: lane holds row n.
// Each half-fragment is a 16-byte contiguous, 16-byte-aligned run -> b128 load.
DEVINL v16h load_frag(const _Float16* src, int ld, int k0, int lane) {
  const _Float16* p = src + (size_t)(lane & 15) * ld + k0 + ((lane & 16) ? 8 : 0);
  const v8h lo = *reinterpret_cast<const v8h*>(p);
  const v8h hi = *reinterpret_cast<const v8h*>(p + 16);
  v16h r;
#pragma unroll
  for (int e = 0; e < 8; ++e) { r[e] = lo[e]; r[e + 8] = hi[e]; }
  return r;
}

DEVINL v8f wmma16(v16h a, v16h b, v8f c) {
  return __builtin_amdgcn_wmma_f32_16x16x32_f16(
      /*neg_a=*/false, a, /*neg_b=*/false, b,
      /*c_mod=*/(short)0, c, /*reuse_a=*/false, /*reuse_b=*/false);
}

// C/D layout: N = lane&15, row of element i is i + 8*(lane>=16)
DEVINL int cd_row_base(int lane) { return (lane & 16) ? 8 : 0; }

// ---------------- kernel 0: weight prep (f32 -> f16, combine, zero-pad) ------
__global__ __launch_bounds__(256)
void prep_kernel(const float* __restrict__ fc1_w, const float* __restrict__ wih,
                 const float* __restrict__ whh,   const float* __restrict__ e1_w,
                 const float* __restrict__ e1_b,  const float* __restrict__ e2_w,
                 const float* __restrict__ q1_w,  const float* __restrict__ q1_b,
                 const float* __restrict__ q2_w,
                 _Float16* __restrict__ fc1_w16, _Float16* __restrict__ wih16,
                 _Float16* __restrict__ whh16,   _Float16* __restrict__ e1c16,
                 float* __restrict__ e1b_pad,    _Float16* __restrict__ e2_16,
                 _Float16* __restrict__ q1p16,   float* __restrict__ q1b_pad,
                 _Float16* __restrict__ q2p16) {
  const int t0 = threadIdx.x + blockIdx.x * blockDim.x;
  const int st = blockDim.x * gridDim.x;
  for (int i = t0; i < HH * OBS; i += st) fc1_w16[i] = (_Float16)fc1_w[i];
  for (int i = t0; i < 3 * HH * HH; i += st) {
    wih16[i] = (_Float16)wih[i];
    whh16[i] = (_Float16)whh[i];
  }
  // combined GNN first layer: rows 0..95 = Wa-Wb, rows 96..191 = Wb (N x K = 192 x 64)
  for (int i = t0; i < 192 * HH; i += st) {
    const int n = i >> 6, k = i & 63;
    float v = (n < EHID) ? (e1_w[(size_t)n * 128 + k] - e1_w[(size_t)n * 128 + 64 + k])
                         : e1_w[(size_t)(n - EHID) * 128 + 64 + k];
    e1c16[i] = (_Float16)v;
  }
  for (int i = t0; i < 192; i += st) e1b_pad[i] = (i < EHID) ? e1_b[i] : 0.f;
  for (int i = t0; i < HH * EHID; i += st) e2_16[i] = (_Float16)e2_w[i];
  // q1 padded to 112 x 128 (zero rows 104..111)
  for (int i = t0; i < QPAD * 128; i += st) {
    const int n = i >> 7;
    q1p16[i] = (_Float16)((n < QHID) ? q1_w[i] : 0.f);
  }
  for (int i = t0; i < QPAD; i += st) q1b_pad[i] = (i < QHID) ? q1_b[i] : 0.f;
  // q2 padded to 16 x 128 (zero cols 104..127)
  for (int i = t0; i < NACT * 128; i += st) {
    const int n = i >> 7, k = i & 127;
    q2p16[i] = (_Float16)((k < QHID) ? q2_w[(size_t)n * QHID + k] : 0.f);
  }
}

// ---------------- fused kernel: one block = one batch (32 agents) ------------
// fc1 -> GRU -> pre(split concat-linear) -> edge relu-mean -> gnn -> q1 -> q2
// 8 waves; every intermediate lives in LDS (114 KB of the 320 KB WGP pool).
__global__ __launch_bounds__(256)
void fused_kernel(const float* __restrict__ inputs, const float* __restrict__ hidden,
                  const _Float16* __restrict__ fc1_w16, const float* __restrict__ fc1_b,
                  const _Float16* __restrict__ wih16, const _Float16* __restrict__ whh16,
                  const float* __restrict__ bih, const float* __restrict__ bhh,
                  const _Float16* __restrict__ e1c16, const float* __restrict__ e1b_pad,
                  const _Float16* __restrict__ e2_16, const float* __restrict__ e2_b,
                  const _Float16* __restrict__ q1p16, const float* __restrict__ q1b_pad,
                  const _Float16* __restrict__ q2p16, const float* __restrict__ q2_b,
                  float* __restrict__ h_out, float* __restrict__ qvals) {
  __shared__ __align__(16) _Float16 s_obs[AA * OBS];   // 8 KB
  __shared__ __align__(16) _Float16 s_hin[AA * HH];    // 4 KB
  __shared__ __align__(16) _Float16 s_x[AA * HH];      // 4 KB
  __shared__ float    s_gx[AA * 192];                  // 24 KB
  __shared__ float    s_gh[AA * 192];                  // 24 KB
  __shared__ __align__(16) _Float16 s_hf[AA * HH];     // 4 KB  (h as f16)
  __shared__ float    s_pre[AA * 192];                 // 24 KB ([pre_a | pre_b])
  __shared__ __align__(16) _Float16 s_mean[AA * EHID]; // 6 KB
  __shared__ __align__(16) _Float16 s_emb[AA * 128];   // 8 KB  ([h | gnn])
  __shared__ __align__(16) _Float16 s_hid[AA * 128];   // 8 KB  (relu(q1), K-pad)

  const int tid   = threadIdx.x;
  const int lane  = tid & 31;
  const int wv    = tid >> 5;            // 8 waves
  const int node0 = blockIdx.x * AA;     // 32 agents per batch

  // ---- stage 0: stage obs (32x128) and hidden (32x64) into LDS as f16 ------
  {
    const float4* src = reinterpret_cast<const float4*>(inputs + (size_t)node0 * OBS);
    for (int i = tid; i < AA * OBS / 4; i += 256) {
      const float4 v = src[i];
      const int b = i * 4;
      s_obs[b + 0] = (_Float16)v.x; s_obs[b + 1] = (_Float16)v.y;
      s_obs[b + 2] = (_Float16)v.z; s_obs[b + 3] = (_Float16)v.w;
    }
    const float4* hsrc = reinterpret_cast<const float4*>(hidden + (size_t)node0 * HH);
    for (int i = tid; i < AA * HH / 4; i += 256) {
      const float4 v = hsrc[i];
      const int b = i * 4;
      s_hin[b + 0] = (_Float16)v.x; s_hin[b + 1] = (_Float16)v.y;
      s_hin[b + 2] = (_Float16)v.z; s_hin[b + 3] = (_Float16)v.w;
    }
  }
  __syncthreads();

  // ---- stage 1: x = relu(obs @ fc1^T + b)  (2 mt x 4 nt = 8 tiles) ---------
  for (int t = wv; t < 8; t += 8) {
    const int mt = t >> 2, nt = t & 3;
    const _Float16* Asrc = s_obs + mt * 16 * OBS;
    const _Float16* Wb   = fc1_w16 + (size_t)(nt * 16) * OBS;
    v8f acc = {};
#pragma unroll
    for (int kk = 0; kk < OBS; kk += 32)
      acc = wmma16(load_frag(Asrc, OBS, kk, lane), load_frag(Wb, OBS, kk, lane), acc);
    const int   n    = nt * 16 + (lane & 15);
    const float bias = fc1_b[n];
    const int   mb   = mt * 16 + cd_row_base(lane);
#pragma unroll
    for (int i = 0; i < 8; ++i) {
      float v = acc[i] + bias;
      s_x[(mb + i) * HH + n] = (_Float16)(v > 0.f ? v : 0.f);
    }
  }
  __syncthreads();

  // ---- stage 2: gx = x @ wih^T + bih ; gh = hin @ whh^T + bhh (48 tiles) ---
  for (int t = wv; t < 48; t += 8) {
    const bool isg = (t < 24);
    const int  r   = isg ? t : t - 24;
    const int  mt  = r / 12, nt = r % 12;
    const _Float16* Asrc = (isg ? s_x : s_hin) + mt * 16 * HH;
    const _Float16* Wb   = (isg ? wih16 : whh16) + (size_t)(nt * 16) * HH;
    const float*    Bv   = isg ? bih : bhh;
    float*          Out  = isg ? s_gx : s_gh;
    v8f acc = {};
#pragma unroll
    for (int kk = 0; kk < HH; kk += 32)
      acc = wmma16(load_frag(Asrc, HH, kk, lane), load_frag(Wb, HH, kk, lane), acc);
    const int   n    = nt * 16 + (lane & 15);
    const float bias = Bv[n];
    const int   mb   = mt * 16 + cd_row_base(lane);
#pragma unroll
    for (int i = 0; i < 8; ++i) Out[(mb + i) * 192 + n] = acc[i] + bias;
  }
  __syncthreads();

  // ---- stage 3: GRU gates (fp32), h -> global fp32 + LDS f16 ---------------
  for (int i = tid; i < AA * HH; i += 256) {
    const int m = i >> 6, c = i & 63;
    const float xr = s_gx[m * 192 + c],       hr = s_gh[m * 192 + c];
    const float xz = s_gx[m * 192 + 64 + c],  hz = s_gh[m * 192 + 64 + c];
    const float xn = s_gx[m * 192 + 128 + c], hn = s_gh[m * 192 + 128 + c];
    const float r  = 1.f / (1.f + __expf(-(xr + hr)));
    const float z  = 1.f / (1.f + __expf(-(xz + hz)));
    const float nn = tanhf(xn + r * hn);
    const float hi = hidden[(size_t)node0 * HH + i];
    const float h  = (1.f - z) * nn + z * hi;
    h_out[(size_t)node0 * HH + i] = h;
    s_hf[i] = (_Float16)h;
  }
  __syncthreads();

  // ---- stage 4: pre = h @ [(Wa-Wb)^T | Wb^T] + [e1_b | 0]  (24 tiles) ------
  for (int t = wv; t < 24; t += 8) {
    const int mt = t / 12, nt = t % 12;
    const _Float16* Asrc = s_hf + mt * 16 * HH;
    const _Float16* Wb   = e1c16 + (size_t)(nt * 16) * HH;
    v8f acc = {};
#pragma unroll
    for (int kk = 0; kk < HH; kk += 32)
      acc = wmma16(load_frag(Asrc, HH, kk, lane), load_frag(Wb, HH, kk, lane), acc);
    const int   n    = nt * 16 + (lane & 15);
    const float bias = e1b_pad[n];
    const int   mb   = mt * 16 + cd_row_base(lane);
#pragma unroll
    for (int i = 0; i < 8; ++i) s_pre[(mb + i) * 192 + n] = acc[i] + bias;
  }
  __syncthreads();

  // ---- stage 5: mean_j relu(pre_a[i] + pre_b[j])  (the O(A^2) part) --------
  {
    const int i  = tid >> 3;          // agent 0..31
    const int c0 = (tid & 7) * 12;    // 12-channel slice (8*12 = 96)
    float pa[12], acc[12];
#pragma unroll
    for (int p = 0; p < 12; ++p) { pa[p] = s_pre[i * 192 + c0 + p]; acc[p] = 0.f; }
    for (int j = 0; j < AA; ++j) {
#pragma unroll
      for (int p = 0; p < 12; ++p) {
        const float v = pa[p] + s_pre[j * 192 + EHID + c0 + p];
        acc[p] += (v > 0.f ? v : 0.f);
      }
    }
#pragma unroll
    for (int p = 0; p < 12; ++p)
      s_mean[i * EHID + c0 + p] = (_Float16)(acc[p] * (1.0f / AA));
  }
  __syncthreads();

  // ---- stage 6: emb = [h | mean @ e2^T + e2_b]  (8 gnn tiles + h copy) -----
  {
    const int m = tid >> 3, c8 = tid & 7;   // 256 threads = 32 rows x 8 v8h
    *reinterpret_cast<v8h*>(&s_emb[m * 128 + c8 * 8]) =
        *reinterpret_cast<const v8h*>(&s_hf[m * HH + c8 * 8]);
  }
  // zero-fill K-pad cols 112..127 of s_hid (for the q2 GEMM)
  for (int i = tid; i < AA * 16; i += 256) {
    const int m = i >> 4, c = i & 15;
    s_hid[m * 128 + QPAD + c] = (_Float16)0.f;
  }
  for (int t = wv; t < 8; t += 8) {
    const int mt = t >> 2, nt = t & 3;
    const _Float16* Asrc = s_mean + mt * 16 * EHID;
    const _Float16* Wb   = e2_16 + (size_t)(nt * 16) * EHID;
    v8f acc = {};
#pragma unroll
    for (int kk = 0; kk < EHID; kk += 32)
      acc = wmma16(load_frag(Asrc, EHID, kk, lane), load_frag(Wb, EHID, kk, lane), acc);
    const int   n    = nt * 16 + (lane & 15);
    const float bias = e2_b[n];
    const int   mb   = mt * 16 + cd_row_base(lane);
#pragma unroll
    for (int i = 0; i < 8; ++i)
      s_emb[(mb + i) * 128 + HH + n] = (_Float16)(acc[i] + bias);
  }
  __syncthreads();

  // ---- stage 7: hid = relu(emb @ q1p^T + q1b)  (2 mt x 7 nt = 14 tiles) ----
  for (int t = wv; t < 14; t += 8) {
    const int mt = t / 7, nt = t % 7;
    const _Float16* Asrc = s_emb + mt * 16 * 128;
    const _Float16* Wb   = q1p16 + (size_t)(nt * 16) * 128;
    v8f acc = {};
#pragma unroll
    for (int kk = 0; kk < 128; kk += 32)
      acc = wmma16(load_frag(Asrc, 128, kk, lane), load_frag(Wb, 128, kk, lane), acc);
    const int   n    = nt * 16 + (lane & 15);
    const float bias = q1b_pad[n];
    const int   mb   = mt * 16 + cd_row_base(lane);
#pragma unroll
    for (int i = 0; i < 8; ++i) {
      float v = acc[i] + bias;
      s_hid[(mb + i) * 128 + n] = (_Float16)(v > 0.f ? v : 0.f);
    }
  }
  __syncthreads();

  // ---- stage 8: q = hid @ q2p^T + q2_b  (2 tiles; wave-uniform) ------------
  if (wv < 2) {
    const _Float16* Asrc = s_hid + wv * 16 * 128;
    v8f acc = {};
#pragma unroll
    for (int kk = 0; kk < 128; kk += 32)
      acc = wmma16(load_frag(Asrc, 128, kk, lane), load_frag(q2p16, 128, kk, lane), acc);
    const int   n    = lane & 15;
    const float bias = q2_b[n];
    const int   mb   = wv * 16 + cd_row_base(lane);
#pragma unroll
    for (int i = 0; i < 8; ++i)
      qvals[(size_t)(node0 + mb + i) * NACT + n] = acc[i] + bias;
  }
}

// ---------------- launch -----------------------------------------------------
extern "C" void kernel_launch(void* const* d_in, const int* in_sizes, int n_in,
                              void* d_out, int out_size, void* d_ws, size_t ws_size,
                              hipStream_t stream) {
  const float* inputs = (const float*)d_in[0];
  const float* hidden = (const float*)d_in[1];
  const float* fc1_w  = (const float*)d_in[2];
  const float* fc1_b  = (const float*)d_in[3];
  const float* wih    = (const float*)d_in[4];
  const float* whh    = (const float*)d_in[5];
  const float* bih    = (const float*)d_in[6];
  const float* bhh    = (const float*)d_in[7];
  const float* e1_w   = (const float*)d_in[8];
  const float* e1_b   = (const float*)d_in[9];
  const float* e2_w   = (const float*)d_in[10];
  const float* e2_b   = (const float*)d_in[11];
  const float* q1_w   = (const float*)d_in[12];
  const float* q1_b   = (const float*)d_in[13];
  const float* q2_w   = (const float*)d_in[14];
  const float* q2_b   = (const float*)d_in[15];

  float* qvals = (float*)d_out;                        // B*A*NACT = 262144
  float* h_out = (float*)d_out + (size_t)NODES * NACT; // B*A*H

  char* ws = (char*)d_ws;
  size_t off = 0;
  _Float16* fc1_w16 = (_Float16*)(ws + off); off += (size_t)HH * OBS * 2;
  _Float16* wih16   = (_Float16*)(ws + off); off += (size_t)3 * HH * HH * 2;
  _Float16* whh16   = (_Float16*)(ws + off); off += (size_t)3 * HH * HH * 2;
  _Float16* e1c16   = (_Float16*)(ws + off); off += (size_t)192 * HH * 2;
  _Float16* e2_16   = (_Float16*)(ws + off); off += (size_t)HH * EHID * 2;
  _Float16* q1p16   = (_Float16*)(ws + off); off += (size_t)QPAD * 128 * 2;
  _Float16* q2p16   = (_Float16*)(ws + off); off += (size_t)NACT * 128 * 2;
  float*    e1b_pad = (float*)   (ws + off); off += 192 * 4;
  float*    q1b_pad = (float*)   (ws + off); off += 512;

  prep_kernel<<<32, 256, 0, stream>>>(fc1_w, wih, whh, e1_w, e1_b, e2_w, q1_w, q1_b, q2_w,
                                      fc1_w16, wih16, whh16, e1c16, e1b_pad,
                                      e2_16, q1p16, q1b_pad, q2p16);
  fused_kernel<<<BB, 256, 0, stream>>>(inputs, hidden, fc1_w16, fc1_b,
                                       wih16, whh16, bih, bhh,
                                       e1c16, e1b_pad, e2_16, e2_b,
                                       q1p16, q1b_pad, q2p16, q2_b,
                                       h_out, qvals);
}
